// SwinTransformerBlock_78400333021344
// MI455X (gfx1250) — matrix operand order repaired and checked by reading, not compiled, not run on hardware
//
#include <hip/hip_runtime.h>
#include <hip/hip_bf16.h>
#include <cmath>

typedef __bf16 bf16_t;
typedef __attribute__((ext_vector_type(16))) bf16_t v16bf;
typedef __attribute__((ext_vector_type(8)))  bf16_t v8bf;
typedef __attribute__((ext_vector_type(8)))  float  v8f;

#define WMMA_BF16(a, b, c) \
  __builtin_amdgcn_wmma_f32_16x16x32_bf16(false, (a), false, (b), (short)0, (c), false, false)

static __device__ __forceinline__ v16bf cat8(v8bf lo, v8bf hi) {
  return __builtin_shufflevector(lo, hi, 0,1,2,3,4,5,6,7,8,9,10,11,12,13,14,15);
}

// ---------------------------------------------------------------------------
// Generic bf16 WMMA GEMM:  Y[M,N] = X[M,K] * W[N,K]^T (+bias)(+GELU)(+resid)
// Block: 256 threads (8 waves), tile 128x128, K staged by 64 through LDS.
// Software-pipelined: stage k+1 is fetched global->registers while WMMAs
// consume stage k from LDS; registers are committed to LDS behind the barrier.
// M % 128 == 0, N % 128 == 0, K % 64 == 0 (always true for this model).
// ---------------------------------------------------------------------------
__global__ __launch_bounds__(256) void gemm_bf16(
    const bf16_t* __restrict__ A, const bf16_t* __restrict__ Bw,
    const float* __restrict__ bias, const float* __restrict__ resid,
    float* __restrict__ outF, bf16_t* __restrict__ outB,
    int M, int N, int K, int doGelu)
{
  __shared__ bf16_t As[128][72];   // [m][k], 8-elem pad
  __shared__ bf16_t Bs[128][72];   // [n][k], 8-elem pad

  const int tid  = threadIdx.x;
  const int wave = tid >> 5;
  const int lane = tid & 31;
  const int lm   = lane & 15;
  const int lh   = lane >> 4;
  const int wm   = wave & 3;       // 4 waves along M (32 rows each)
  const int wn   = wave >> 2;      // 2 waves along N (64 cols each)
  const long mBase = (long)blockIdx.y * 128;
  const long nBase = (long)blockIdx.x * 128;

  // per-thread staging coordinates: 4 chunks of 8 bf16 for each of A,B
  int srow[4], scol[4];
  #pragma unroll
  for (int i = 0; i < 4; ++i) {
    const int chunk = tid + i * 256;        // 1024 chunks of 8 bf16
    srow[i] = chunk >> 3;
    scol[i] = (chunk & 7) << 3;
  }

  v8bf pa[4], pb[4];
  auto ldg = [&](int kk) {
    #pragma unroll
    for (int i = 0; i < 4; ++i) {
      pa[i] = *(const v8bf*)(&A [(mBase + srow[i]) * (long)K + kk + scol[i]]);
      pb[i] = *(const v8bf*)(&Bw[(nBase + srow[i]) * (long)K + kk + scol[i]]);
    }
  };
  auto commit = [&]() {
    #pragma unroll
    for (int i = 0; i < 4; ++i) {
      *(v8bf*)(&As[srow[i]][scol[i]]) = pa[i];
      *(v8bf*)(&Bs[srow[i]][scol[i]]) = pb[i];
    }
  };

  v8f acc[2][4];
  #pragma unroll
  for (int mt = 0; mt < 2; ++mt)
    #pragma unroll
    for (int nt = 0; nt < 4; ++nt)
      #pragma unroll
      for (int r = 0; r < 8; ++r) acc[mt][nt][r] = 0.0f;

  // prologue: stage 0 into LDS
  ldg(0);
  commit();
  __syncthreads();

  for (int k0 = 0; k0 < K; k0 += 64) {
    const bool more = (k0 + 64) < K;
    if (more) ldg(k0 + 64);                       // async wrt compute below
    if (k0 + 128 < K) {                           // warm GL2 two stages ahead
      __builtin_prefetch(&A [(mBase + srow[0]) * (long)K + k0 + 128 + scol[0]], 0, 1);
      __builtin_prefetch(&Bw[(nBase + srow[0]) * (long)K + k0 + 128 + scol[0]], 0, 1);
    }
    #pragma unroll
    for (int ks = 0; ks < 2; ++ks) {
      const int kk = ks * 32;
      v16bf af[2];
      #pragma unroll
      for (int mt = 0; mt < 2; ++mt) {
        const int row = wm * 32 + mt * 16 + lm;
        v8bf lo = *(const v8bf*)(&As[row][kk + lh * 16]);       // K 0..7 / 16..23
        v8bf hi = *(const v8bf*)(&As[row][kk + lh * 16 + 8]);   // K 8..15 / 24..31
        af[mt] = cat8(lo, hi);
      }
      v16bf bfr[4];
      #pragma unroll
      for (int nt = 0; nt < 4; ++nt) {
        const int nr = wn * 64 + nt * 16 + lm;
        v8bf lo = *(const v8bf*)(&Bs[nr][kk + lh * 16]);        // K 0..7 / 16..23
        v8bf hi = *(const v8bf*)(&Bs[nr][kk + lh * 16 + 8]);    // K 8..15 / 24..31
        bfr[nt] = cat8(lo, hi);
      }
      #pragma unroll
      for (int mt = 0; mt < 2; ++mt)
        #pragma unroll
        for (int nt = 0; nt < 4; ++nt)
          acc[mt][nt] = WMMA_BF16(af[mt], bfr[nt], acc[mt][nt]);
    }
    __syncthreads();          // everyone done reading stage k
    if (more) commit();       // overwrite LDS with stage k+1
    __syncthreads();          // stage k+1 visible
  }

  #pragma unroll
  for (int nt = 0; nt < 4; ++nt) {
    const long n = nBase + wn * 64 + nt * 16 + lm;
    const float bv = bias ? bias[n] : 0.0f;
    #pragma unroll
    for (int mt = 0; mt < 2; ++mt) {
      #pragma unroll
      for (int r = 0; r < 8; ++r) {
        const long m = mBase + wm * 32 + mt * 16 + lh * 8 + r;  // C-layout: VGPR r -> M = r + 8*half
        float v = acc[mt][nt][r] + bv;
        if (doGelu) v = 0.5f * v * (1.0f + erff(v * 0.70710678118654752f));
        if (resid) v += resid[m * (long)N + n];
        if (outF) outF[m * (long)N + n] = v;
        if (outB) outB[m * (long)N + n] = (bf16_t)v;
      }
    }
  }
}

// ---------------------------------------------------------------------------
// Fused window attention: one block per (head, window, batch).
// L=49 padded to 64, dk=32. Gathers Q/K/V (token-major, bf16) through the
// window->token map (shift folded in), computes softmax(QK^T + rpb (+mask))V
// with WMMA, scatters bf16 output token-major.
// ---------------------------------------------------------------------------
__global__ __launch_bounds__(256) void win_attn(
    const bf16_t* __restrict__ Qg, const bf16_t* __restrict__ Kg,
    const bf16_t* __restrict__ Vg, const float* __restrict__ rpb,
    bf16_t* __restrict__ Og, int shifted, int masked)
{
  __shared__ int    tok[64];
  __shared__ int    lab[64];
  __shared__ bf16_t Qs[64][40];   // [i][d]
  __shared__ bf16_t Ks[64][40];   // [j][d]  (acts as B operand [n][k])
  __shared__ bf16_t Vt[32][72];   // [d][j]  (V transposed -> B operand [n][k])
  __shared__ bf16_t Ps[64][72];   // softmax probs [i][j]
  __shared__ float  Ss[64][64];

  const int head = blockIdx.x;
  const int widx = blockIdx.y;
  const int bb   = blockIdx.z;
  const int tid  = threadIdx.x;
  const int wave = tid >> 5, lane = tid & 31, lm = lane & 15, lh = lane >> 4;

  if (tid < 64) {
    const int l = tid;
    if (l < 49) {
      const int wy = widx >> 3, wx = widx & 7;
      const int r = l / 7, c = l % 7;
      const int ph = wy * 7 + r, pw = wx * 7 + c;       // coords in (shifted) layout
      const int sh = shifted ? (ph + 3) % 56 : ph;      // source coords after roll(-3)
      const int sw = shifted ? (pw + 3) % 56 : pw;
      tok[l] = bb * 3136 + sh * 56 + sw;
      const int rh = (ph < 49) ? 0 : ((ph < 53) ? 1 : 2);
      const int rw = (pw < 49) ? 0 : ((pw < 53) ? 1 : 2);
      lab[l] = rh * 3 + rw;
    } else { tok[l] = -1; lab[l] = 0; }
  }
  __syncthreads();

  { // load Q/K rows (64B per row, 4 chunks)
    const int row = tid >> 2;
    const int ch  = (tid & 3) << 3;
    const int t   = tok[row];
    if (t >= 0) {
      const long base = (long)t * 512 + head * 32 + ch;
      *(v8bf*)(&Qs[row][ch]) = *(const v8bf*)(&Qg[base]);
      *(v8bf*)(&Ks[row][ch]) = *(const v8bf*)(&Kg[base]);
    } else {
      v8bf z;
      #pragma unroll
      for (int i = 0; i < 8; ++i) z[i] = (bf16_t)0.0f;
      *(v8bf*)(&Qs[row][ch]) = z;
      *(v8bf*)(&Ks[row][ch]) = z;
    }
  }
  #pragma unroll
  for (int i = 0; i < 8; ++i) {   // load V transposed
    const int idx = tid + i * 256;
    const int j = idx & 63, d = idx >> 6;
    const int t = tok[j];
    Vt[d][j] = (t >= 0) ? Vg[(long)t * 512 + head * 32 + d] : (bf16_t)0.0f;
  }
  __syncthreads();

  // scores: 16 tiles of 16x16, 2 per wave; K = dk = 32 -> single WMMA each
  #pragma unroll
  for (int p = 0; p < 2; ++p) {
    const int tt = wave + p * 8;
    const int it = tt >> 2, jt = tt & 3;
    v8bf a0 = *(const v8bf*)(&Qs[it * 16 + lm][lh * 16]);
    v8bf a1 = *(const v8bf*)(&Qs[it * 16 + lm][lh * 16 + 8]);
    v16bf af = cat8(a0, a1);
    v8bf b0 = *(const v8bf*)(&Ks[jt * 16 + lm][lh * 16]);
    v8bf b1 = *(const v8bf*)(&Ks[jt * 16 + lm][lh * 16 + 8]);
    v16bf bfr = cat8(b0, b1);
    v8f acc;
    #pragma unroll
    for (int r = 0; r < 8; ++r) acc[r] = 0.0f;
    acc = WMMA_BF16(af, bfr, acc);
    #pragma unroll
    for (int r = 0; r < 8; ++r) {
      const int i = it * 16 + lh * 8 + r;
      const int j = jt * 16 + lm;
      float s = acc[r];
      if (j >= 49) {
        s = -1e30f;                               // padded keys excluded
      } else if (i < 49) {
        const int ri = i / 7, ci = i % 7, rj = j / 7, cj = j % 7;
        const int bidx = (ri - rj + 6) * 13 + (ci - cj + 6);
        s += rpb[bidx * 16 + head];               // rpb[(2*7-1)^2][16] row-major
        if (masked && (lab[i] != lab[j])) s -= 100.0f;
      }
      Ss[i][j] = s;
    }
  }
  __syncthreads();

  if (tid < 64) {  // row softmax (tiny: 64x64)
    const int i = tid;
    float mx = -1e30f;
    for (int j = 0; j < 64; ++j) mx = fmaxf(mx, Ss[i][j]);
    float sum = 0.0f;
    for (int j = 0; j < 64; ++j) sum += expf(Ss[i][j] - mx);
    const float inv = 1.0f / sum;
    for (int j = 0; j < 64; ++j) Ps[i][j] = (bf16_t)(expf(Ss[i][j] - mx) * inv);
  }
  __syncthreads();

  { // out = P @ V : 8 tiles (4 i-tiles x 2 d-tiles), 1 per wave, K=64 -> 2 WMMAs
    const int it = wave >> 1, dt = wave & 1;
    v8f acc;
    #pragma unroll
    for (int r = 0; r < 8; ++r) acc[r] = 0.0f;
    #pragma unroll
    for (int ks = 0; ks < 2; ++ks) {
      const int kk = ks * 32;
      v8bf a0 = *(const v8bf*)(&Ps[it * 16 + lm][kk + lh * 16]);
      v8bf a1 = *(const v8bf*)(&Ps[it * 16 + lm][kk + lh * 16 + 8]);
      v16bf af = cat8(a0, a1);
      v8bf b0 = *(const v8bf*)(&Vt[dt * 16 + lm][kk + lh * 16]);
      v8bf b1 = *(const v8bf*)(&Vt[dt * 16 + lm][kk + lh * 16 + 8]);
      v16bf bfr = cat8(b0, b1);
      acc = WMMA_BF16(af, bfr, acc);
    }
    #pragma unroll
    for (int r = 0; r < 8; ++r) {
      const int i = it * 16 + lh * 8 + r;
      if (i < 49) {
        const int d = dt * 16 + lm;
        Og[(long)tok[i] * 512 + head * 32 + d] = (bf16_t)acc[r];
      }
    }
  }
}

// ---------------------------------------------------------------------------
// LayerNorm over C=512, one block (256 threads) per token row.
// ---------------------------------------------------------------------------
__global__ __launch_bounds__(256) void ln512(
    const float* __restrict__ x, const float* __restrict__ g,
    const float* __restrict__ bta, float* __restrict__ outF,
    bf16_t* __restrict__ outB)
{
  __shared__ float red[256];
  const int tid = threadIdx.x;
  const long row = blockIdx.x;
  const float* xr = x + row * 512;
  const float v0 = xr[tid];
  const float v1 = xr[tid + 256];

  red[tid] = v0 + v1;
  __syncthreads();
  for (int s = 128; s > 0; s >>= 1) {
    if (tid < s) red[tid] += red[tid + s];
    __syncthreads();
  }
  const float mu = red[0] * (1.0f / 512.0f);
  __syncthreads();

  const float d0 = v0 - mu, d1 = v1 - mu;
  red[tid] = d0 * d0 + d1 * d1;
  __syncthreads();
  for (int s = 128; s > 0; s >>= 1) {
    if (tid < s) red[tid] += red[tid + s];
    __syncthreads();
  }
  const float inv = rsqrtf(red[0] * (1.0f / 512.0f) + 1e-5f);

  const float y0 = d0 * inv * g[tid] + bta[tid];
  const float y1 = d1 * inv * g[tid + 256] + bta[tid + 256];
  if (outF) { outF[row * 512 + tid] = y0; outF[row * 512 + tid + 256] = y1; }
  if (outB) { outB[row * 512 + tid] = (bf16_t)y0; outB[row * 512 + tid + 256] = (bf16_t)y1; }
}

__global__ void cvt_f32_bf16(const float* __restrict__ in, bf16_t* __restrict__ out, int n) {
  const int i = blockIdx.x * 256 + threadIdx.x;
  if (i < n) out[i] = (bf16_t)in[i];
}

// ---------------------------------------------------------------------------
extern "C" void kernel_launch(void* const* d_in, const int* in_sizes, int n_in,
                              void* d_out, int out_size, void* d_ws, size_t ws_size,
                              hipStream_t stream)
{
  (void)in_sizes; (void)n_in; (void)out_size; (void)ws_size;
  const int M = 8 * 56 * 56;     // 25088 tokens
  const int C = 512, C4 = 2048;

  const float* x    = (const float*)d_in[0];
  const float* ln1g = (const float*)d_in[1];
  const float* ln1b = (const float*)d_in[2];
  const float* ln2g = (const float*)d_in[3];
  const float* ln2b = (const float*)d_in[4];
  const float* ln3g = (const float*)d_in[5];
  const float* ln3b = (const float*)d_in[6];
  const float* wq1  = (const float*)d_in[7];
  const float* wk1  = (const float*)d_in[8];
  const float* wv1  = (const float*)d_in[9];
  const float* wo1  = (const float*)d_in[10];
  const float* rpb1 = (const float*)d_in[11];
  const float* wq2  = (const float*)d_in[12];
  const float* wk2  = (const float*)d_in[13];
  const float* wv2  = (const float*)d_in[14];
  const float* wo2  = (const float*)d_in[15];
  const float* rpb2 = (const float*)d_in[16];
  const float* m1w1 = (const float*)d_in[17];
  const float* m1b1 = (const float*)d_in[18];
  const float* m1w2 = (const float*)d_in[19];
  const float* m1b2 = (const float*)d_in[20];
  const float* m2w1 = (const float*)d_in[21];
  const float* m2b1 = (const float*)d_in[22];
  const float* m2w2 = (const float*)d_in[23];
  const float* m2b2 = (const float*)d_in[24];

  char* ws = (char*)d_ws;
  size_t off = 0;
  auto take = [&](size_t bytes) -> char* {
    char* p = ws + off;
    off += (bytes + 255) & ~(size_t)255;
    return p;
  };

  bf16_t* Wq1b = (bf16_t*)take((size_t)C * C * 2);
  bf16_t* Wk1b = (bf16_t*)take((size_t)C * C * 2);
  bf16_t* Wv1b = (bf16_t*)take((size_t)C * C * 2);
  bf16_t* Wo1b = (bf16_t*)take((size_t)C * C * 2);
  bf16_t* Wq2b = (bf16_t*)take((size_t)C * C * 2);
  bf16_t* Wk2b = (bf16_t*)take((size_t)C * C * 2);
  bf16_t* Wv2b = (bf16_t*)take((size_t)C * C * 2);
  bf16_t* Wo2b = (bf16_t*)take((size_t)C * C * 2);
  bf16_t* M1w1b = (bf16_t*)take((size_t)C4 * C * 2);
  bf16_t* M1w2b = (bf16_t*)take((size_t)C * C4 * 2);
  bf16_t* M2w1b = (bf16_t*)take((size_t)C4 * C * 2);
  bf16_t* M2w2b = (bf16_t*)take((size_t)C * C4 * 2);
  bf16_t* Abuf = (bf16_t*)take((size_t)M * C * 2);   // LN output (bf16 GEMM input)
  bf16_t* Qb   = (bf16_t*)take((size_t)M * C * 2);
  bf16_t* Kb   = (bf16_t*)take((size_t)M * C * 2);
  bf16_t* Vb   = (bf16_t*)take((size_t)M * C * 2);
  bf16_t* AOb  = (bf16_t*)take((size_t)M * C * 2);   // attention output (pre-wo)
  bf16_t* Mid  = (bf16_t*)take((size_t)M * C4 * 2);  // MLP hidden
  float*  F1   = (float*)take((size_t)M * C * 4);    // x_  (later x3)
  float*  F2   = (float*)take((size_t)M * C * 4);    // x2
  float*  Hb   = (float*)take((size_t)M * C * 4);    // h = LN3(x2), fp32 residual

  auto cvt = [&](const float* src, bf16_t* dst, int n) {
    cvt_f32_bf16<<<(n + 255) / 256, 256, 0, stream>>>(src, dst, n);
  };
  cvt(wq1, Wq1b, C * C);   cvt(wk1, Wk1b, C * C);
  cvt(wv1, Wv1b, C * C);   cvt(wo1, Wo1b, C * C);
  cvt(wq2, Wq2b, C * C);   cvt(wk2, Wk2b, C * C);
  cvt(wv2, Wv2b, C * C);   cvt(wo2, Wo2b, C * C);
  cvt(m1w1, M1w1b, C4 * C); cvt(m1w2, M1w2b, C * C4);
  cvt(m2w1, M2w1b, C4 * C); cvt(m2w2, M2w2b, C * C4);

  const dim3 blk(256);
  const dim3 g512(C / 128, M / 128);
  const dim3 g2048(C4 / 128, M / 128);
  const dim3 gattn(16, 64, 8);   // heads x windows x batch

  // ---- branch 1: plain window attention ----
  ln512<<<M, blk, 0, stream>>>(x, ln1g, ln1b, nullptr, Abuf);
  gemm_bf16<<<g512, blk, 0, stream>>>(Abuf, Wq1b, nullptr, nullptr, nullptr, Qb, M, C, C, 0);
  gemm_bf16<<<g512, blk, 0, stream>>>(Abuf, Wk1b, nullptr, nullptr, nullptr, Kb, M, C, C, 0);
  gemm_bf16<<<g512, blk, 0, stream>>>(Abuf, Wv1b, nullptr, nullptr, nullptr, Vb, M, C, C, 0);
  win_attn<<<gattn, blk, 0, stream>>>(Qb, Kb, Vb, rpb1, AOb, 0, 0);
  gemm_bf16<<<g512, blk, 0, stream>>>(AOb, Wo1b, nullptr, x, F1, nullptr, M, C, C, 0);  // x_ = x + attn

  // ---- MLP 1 ----
  ln512<<<M, blk, 0, stream>>>(F1, ln2g, ln2b, nullptr, Abuf);
  gemm_bf16<<<g2048, blk, 0, stream>>>(Abuf, M1w1b, m1b1, nullptr, nullptr, Mid, M, C4, C, 1);
  gemm_bf16<<<g512, blk, 0, stream>>>(Mid, M1w2b, m1b2, F1, F2, nullptr, M, C, C4, 0);  // x2

  // ---- branch 2: shifted window attention (residual = h = LN3(x2)) ----
  ln512<<<M, blk, 0, stream>>>(F2, ln3g, ln3b, Hb, Abuf);
  gemm_bf16<<<g512, blk, 0, stream>>>(Abuf, Wq2b, nullptr, nullptr, nullptr, Qb, M, C, C, 0);
  gemm_bf16<<<g512, blk, 0, stream>>>(Abuf, Wk2b, nullptr, nullptr, nullptr, Kb, M, C, C, 0);
  gemm_bf16<<<g512, blk, 0, stream>>>(Abuf, Wv2b, nullptr, nullptr, nullptr, Vb, M, C, C, 0);
  win_attn<<<gattn, blk, 0, stream>>>(Qb, Kb, Vb, rpb2, AOb, 1, 1);
  gemm_bf16<<<g512, blk, 0, stream>>>(AOb, Wo2b, nullptr, Hb, F1, nullptr, M, C, C, 0); // x3

  // ---- MLP 2 (reference reuses ln3 params) ----
  ln512<<<M, blk, 0, stream>>>(F1, ln3g, ln3b, nullptr, Abuf);
  gemm_bf16<<<g2048, blk, 0, stream>>>(Abuf, M2w1b, m2b1, nullptr, nullptr, Mid, M, C4, C, 1);
  gemm_bf16<<<g512, blk, 0, stream>>>(Mid, M2w2b, m2b2, F1, (float*)d_out, nullptr, M, C, C4, 0);
}
